// MultiheadAttention_58153857188034
// MI455X (gfx1250) — matrix operand order repaired and checked by reading, not compiled, not run on hardware
//
#include <hip/hip_runtime.h>
#include <hip/hip_bf16.h>

// ---------------------------------------------------------------------------
// Sinkhorn Multihead Attention for MI455X (gfx1250, wave32, WMMA).
// L=1024, B=8, E=1024, H=16, D=64.  All GEMMs via v_wmma_f32_16x16x32_f16,
// 2x2 register blocking per wave; big GEMMs use double-buffered LDS staging
// with CDNA5 async global->LDS loads (ASYNCcnt) when the builtin is available.
// ---------------------------------------------------------------------------

typedef __attribute__((ext_vector_type(16))) _Float16 v16h;
typedef __attribute__((ext_vector_type(8)))  _Float16 v8h;
typedef __attribute__((ext_vector_type(8)))  float    v8f;
typedef __attribute__((ext_vector_type(4)))  float    v4f;

#define SL 1024   // sequence length
#define SB 8      // batch
#define SE 1024   // embed
#define SH 16     // heads
#define SD 64     // head dim
#define SM (SL * SB)   // 8192 GEMM rows

// CDNA5 async global->LDS path (ASYNCcnt), guarded so either toolchain builds.
#if defined(__HIP_DEVICE_COMPILE__) && defined(__gfx1250__)
#  if __has_builtin(__builtin_amdgcn_global_load_async_to_lds_b128) && \
      __has_builtin(__builtin_amdgcn_s_wait_asynccnt)
#    define USE_ASYNC_LDS 1
#  endif
#endif
#ifndef USE_ASYNC_LDS
#  define USE_ASYNC_LDS 0
#endif

// Builtin parameter type is 'int __attribute__((vector_size(16)))*' with
// addrspace(1) for the global source and addrspace(3) for the LDS dest.
typedef int v4i_raw __attribute__((vector_size(16)));
typedef __attribute__((address_space(1))) v4i_raw GV4i;
typedef __attribute__((address_space(3))) v4i_raw LV4i;

#define LDS_PAD 40   // halves per 32-half tile row (80B: 16B-aligned, staggers banks)

// ---------------------------------------------------------------------------
// WMMA fragment loaders (wave32 layouts per CDNA5 ISA 7.12.2)
// ---------------------------------------------------------------------------

// A 16x32 f16 fragment, row-major src, ld = K stride.  Lane group g holds
// K = {g*8..g*8+7} in elems 0..7 and K = {16+g*8..} in elems 8..15.
__device__ __forceinline__ v16h load_fragA_f16(const _Float16* __restrict__ base,
                                               int ld, int r0, int k0, int lane) {
    int g = lane >> 4;
    int r = lane & 15;
    const _Float16* p = base + (size_t)(r0 + r) * ld + k0 + g * 8;
    v8h lo = *(const v8h*)(p);
    v8h hi = *(const v8h*)(p + 16);
    v16h out;
#pragma unroll
    for (int i = 0; i < 8; ++i) { out[i] = lo[i]; out[i + 8] = hi[i]; }
    return out;
}

// Same A layout, f32 source (Sinkhorn probabilities) with inline convert.
__device__ __forceinline__ v16h load_fragA_f32src(const float* __restrict__ base,
                                                  int ld, int r0, int k0, int lane) {
    int g = lane >> 4;
    int r = lane & 15;
    const float* p = base + (size_t)(r0 + r) * ld + k0 + g * 8;
    v4f a0 = *(const v4f*)(p);
    v4f a1 = *(const v4f*)(p + 4);
    v4f b0 = *(const v4f*)(p + 16);
    v4f b1 = *(const v4f*)(p + 20);
    v16h out;
#pragma unroll
    for (int i = 0; i < 4; ++i) {
        out[i]      = (_Float16)a0[i];
        out[i + 4]  = (_Float16)a1[i];
        out[i + 8]  = (_Float16)b0[i];
        out[i + 12] = (_Float16)b1[i];
    }
    return out;
}

// B 32x16 f16 fragment.  Source "Bt": row n of Bt = column n of B, ld = K
// stride.  Lane group g holds K = {g*8..g*8+7} in elems 0..7, K = {16+g*8..} in 8..15.
__device__ __forceinline__ v16h load_fragB_f16(const _Float16* __restrict__ baseT,
                                               int ld, int c0, int k0, int lane) {
    int g = lane >> 4;
    int c = lane & 15;
    const _Float16* p = baseT + (size_t)(c0 + c) * ld + k0 + g * 8;
    v8h lo = *(const v8h*)(p);
    v8h hi = *(const v8h*)(p + 16);
    v16h out;
#pragma unroll
    for (int i = 0; i < 8; ++i) { out[i] = lo[i]; out[i + 8] = hi[i]; }
    return out;
}

// LDS-tile variants (tile is exactly 32 halves wide, padded rows).
__device__ __forceinline__ v16h lds_fragA(const _Float16* t, int r0, int lane) {
    int g = lane >> 4;
    int r = lane & 15;
    const _Float16* p = t + (r0 + r) * LDS_PAD + g * 8;
    v8h lo = *(const v8h*)(p);
    v8h hi = *(const v8h*)(p + 16);
    v16h out;
#pragma unroll
    for (int i = 0; i < 8; ++i) { out[i] = lo[i]; out[i + 8] = hi[i]; }
    return out;
}

__device__ __forceinline__ v16h lds_fragB(const _Float16* t, int c0, int lane) {
    int g = lane >> 4;
    int c = lane & 15;
    const _Float16* p = t + (c0 + c) * LDS_PAD + g * 8;
    v8h lo = *(const v8h*)(p);
    v8h hi = *(const v8h*)(p + 16);
    v16h out;
#pragma unroll
    for (int i = 0; i < 8; ++i) { out[i] = lo[i]; out[i + 8] = hi[i]; }
    return out;
}

#define WMMA_F16(a, b, c) \
    __builtin_amdgcn_wmma_f32_16x16x32_f16(false, (a), false, (b), (short)0, (c), false, false)

// ---------------------------------------------------------------------------
// Global -> LDS staging: one 16B chunk per call.
// ---------------------------------------------------------------------------
__device__ __forceinline__ void stage_chunk(const _Float16* gsrc, _Float16* ldst) {
#if USE_ASYNC_LDS
    __builtin_amdgcn_global_load_async_to_lds_b128(
        (GV4i*)(size_t)gsrc, (LV4i*)(unsigned)(size_t)ldst, 0, 0);
#else
    *(v8h*)ldst = *(const v8h*)gsrc;
#endif
}

// Stage one k-step: A tile 128x32 (512 chunks -> 2/thread), B tile 64x32
// (256 chunks -> 1/thread).  3 staging ops per thread per step.
__device__ __forceinline__ void stage_AB(const _Float16* __restrict__ A,
                                         const _Float16* __restrict__ Bt,
                                         int lda, int ldb, int mblk, int nblk,
                                         int k, _Float16* bufA, _Float16* bufB,
                                         int tid) {
    {
        int row = tid >> 2, ch = tid & 3;
        stage_chunk(A + (size_t)(mblk + row) * lda + k + ch * 8,
                    bufA + row * LDS_PAD + ch * 8);
    }
    {
        int ca = tid + 256;
        int row = ca >> 2, ch = ca & 3;
        stage_chunk(A + (size_t)(mblk + row) * lda + k + ch * 8,
                    bufA + row * LDS_PAD + ch * 8);
    }
    {
        int row = tid >> 2, ch = tid & 3;
        stage_chunk(Bt + (size_t)(nblk + row) * ldb + k + ch * 8,
                    bufB + row * LDS_PAD + ch * 8);
    }
}

// ---------------------------------------------------------------------------
// Double-buffered LDS-staged GEMM core: block tile 128(M) x 64(N), 8 waves
// arranged 4x2, each wave owns a 32x32 tile computed as 2x2 WMMA blocks.
// ---------------------------------------------------------------------------
__device__ __forceinline__ void gemm_core_staged(
    const _Float16* __restrict__ A, const _Float16* __restrict__ Bt,
    int lda, int ldb, int K, int mblk, int nblk,
    _Float16 (*ldsA)[128 * LDS_PAD], _Float16 (*ldsB)[64 * LDS_PAD],
    v8f acc[2][2]) {
    int tid = threadIdx.x;
    int lane = tid & 31, wv = tid >> 5;
    int m0 = (wv >> 1) * 32;
    int n0 = (wv & 1) * 32;
    int nk = K / 32;

    stage_AB(A, Bt, lda, ldb, mblk, nblk, 0, ldsA[0], ldsB[0], tid);
    for (int ki = 0; ki < nk; ++ki) {
        int cur = ki & 1;
        bool more = (ki + 1 < nk);
        if (more)
            stage_AB(A, Bt, lda, ldb, mblk, nblk, (ki + 1) * 32,
                     ldsA[cur ^ 1], ldsB[cur ^ 1], tid);
#if USE_ASYNC_LDS
        if (more) __builtin_amdgcn_s_wait_asynccnt(3);   // prior step's 3 ops done
        else      __builtin_amdgcn_s_wait_asynccnt(0);
#endif
        __syncthreads();
        v16h a0 = lds_fragA(ldsA[cur], m0, lane);
        v16h a1 = lds_fragA(ldsA[cur], m0 + 16, lane);
        v16h b0 = lds_fragB(ldsB[cur], n0, lane);
        v16h b1 = lds_fragB(ldsB[cur], n0 + 16, lane);
        acc[0][0] = WMMA_F16(a0, b0, acc[0][0]);
        acc[0][1] = WMMA_F16(a0, b1, acc[0][1]);
        acc[1][0] = WMMA_F16(a1, b0, acc[1][0]);
        acc[1][1] = WMMA_F16(a1, b1, acc[1][1]);
        __syncthreads();
    }
}

// ---------------------------------------------------------------------------
// Stage 0: fp32 -> fp16 conversion (x, in_proj_weight, out_proj_weight)
// ---------------------------------------------------------------------------
__global__ void cvt_f32_to_f16(const float* __restrict__ src,
                               _Float16* __restrict__ dst, int n) {
    int i = blockIdx.x * blockDim.x + threadIdx.x;
    int stride = gridDim.x * blockDim.x;
    for (; i < n; i += stride) dst[i] = (_Float16)src[i];
}

// ---------------------------------------------------------------------------
// Stage 1: QKV projection, LDS-staged.  C[8192,3072] = xh @ wh^T + bias,
// scattered into qh[b,h,l,d], kh[b,h,l,d], vt[b,h,d,l] (v transposed so the
// AV GEMM's B loader sees contiguous K runs).
// ---------------------------------------------------------------------------
__global__ void __launch_bounds__(256) qkv_gemm(
    const _Float16* __restrict__ xh, const _Float16* __restrict__ wh,
    const float* __restrict__ bias, _Float16* __restrict__ qh,
    _Float16* __restrict__ kh, _Float16* __restrict__ vt) {
    __shared__ _Float16 ldsA[2][128 * LDS_PAD];
    __shared__ _Float16 ldsB[2][64 * LDS_PAD];
    int mblk = blockIdx.y * 128;
    int nblk = blockIdx.x * 64;
    v8f acc[2][2] = {};
    gemm_core_staged(xh, wh, SE, SE, SE, mblk, nblk, ldsA, ldsB, acc);

    int lane = threadIdx.x & 31, wv = threadIdx.x >> 5;
    int g = lane >> 4, c = lane & 15;
    int mw = mblk + (wv >> 1) * 32;
    int nw = nblk + (wv & 1) * 32;
#pragma unroll
    for (int tm = 0; tm < 2; ++tm)
#pragma unroll
        for (int tn = 0; tn < 2; ++tn) {
            int n = nw + tn * 16 + c;
            int which = n >> 10;        // 0=q 1=k 2=v (uniform per sub-tile)
            int e = n & (SE - 1);
            int h = e >> 6;
            int d = e & (SD - 1);
            float bj = bias[n];
#pragma unroll
            for (int i = 0; i < 8; ++i) {
                int m = mw + tm * 16 + g * 8 + i;  // row = l*B + b
                int l = m >> 3, bb = m & 7;
                int head = bb * SH + h;
                _Float16 v = (_Float16)(acc[tm][tn][i] + bj);
                if (which == 0)      qh[((size_t)head * SL + l) * SD + d] = v;
                else if (which == 1) kh[((size_t)head * SL + l) * SD + d] = v;
                else                 vt[((size_t)head * SD + d) * SL + l] = v;
            }
        }
}

// ---------------------------------------------------------------------------
// Stage 2: scores.  S[head,l,s] = (q[l,:].k[s,:])/sqrt(D).  K=64 (2 steps),
// direct global fragment loads, 2x2 register blocking.
// ---------------------------------------------------------------------------
__global__ void __launch_bounds__(256) scores_gemm(
    const _Float16* __restrict__ qh, const _Float16* __restrict__ kh,
    float* __restrict__ S) {
    int lane = threadIdx.x & 31, wv = threadIdx.x >> 5;
    int head = blockIdx.z;
    int m0 = blockIdx.y * 128 + (wv >> 1) * 32;
    int n0 = blockIdx.x * 64 + (wv & 1) * 32;
    const _Float16* qb = qh + (size_t)head * SL * SD;
    const _Float16* kb = kh + (size_t)head * SL * SD;
    v8f acc[2][2] = {};
#pragma unroll
    for (int k = 0; k < SD; k += 32) {
        v16h a0 = load_fragA_f16(qb, SD, m0, k, lane);
        v16h a1 = load_fragA_f16(qb, SD, m0 + 16, k, lane);
        v16h b0 = load_fragB_f16(kb, SD, n0, k, lane);
        v16h b1 = load_fragB_f16(kb, SD, n0 + 16, k, lane);
        acc[0][0] = WMMA_F16(a0, b0, acc[0][0]);
        acc[0][1] = WMMA_F16(a0, b1, acc[0][1]);
        acc[1][0] = WMMA_F16(a1, b0, acc[1][0]);
        acc[1][1] = WMMA_F16(a1, b1, acc[1][1]);
    }
    float* Sh = S + (size_t)head * SL * SL;
    int g = lane >> 4, c = lane & 15;
    const float scale = 0.125f;   // 1/sqrt(64)
#pragma unroll
    for (int tm = 0; tm < 2; ++tm)
#pragma unroll
        for (int tn = 0; tn < 2; ++tn)
#pragma unroll
            for (int i = 0; i < 8; ++i) {
                int m = m0 + tm * 16 + g * 8 + i;
                int n = n0 + tn * 16 + c;
                Sh[(size_t)m * SL + n] = acc[tm][tn][i] * scale;
            }
}

// ---------------------------------------------------------------------------
// Stage 3a: column logsumexp (axis=-2), online single pass, coalesced.
// ---------------------------------------------------------------------------
__global__ void sinkhorn_col(const float* __restrict__ S,
                             float* __restrict__ colLSE) {
    int head = blockIdx.y;
    int s = blockIdx.x * blockDim.x + threadIdx.x;
    const float* Sh = S + (size_t)head * SL * SL;
    float m = -1e30f, ssum = 0.0f;
    for (int l = 0; l < SL; ++l) {
        float v = Sh[(size_t)l * SL + s];
        float nm = fmaxf(m, v);
        ssum = ssum * __expf(m - nm) + __expf(v - nm);
        m = nm;
    }
    colLSE[head * SL + s] = m + __logf(ssum);
}

// ---------------------------------------------------------------------------
// Stage 3b: fused column-subtract + row LSE + row-subtract (LDS tree
// reduction).  Final iteration writes exp(f) = attention probabilities.
// ---------------------------------------------------------------------------
__global__ void sinkhorn_row(float* __restrict__ S,
                             const float* __restrict__ colLSE,
                             int final_exp) {
    int head = blockIdx.y;
    int l = blockIdx.x;
    float* Sh = S + ((size_t)head * SL + l) * SL;
    const float* cl = colLSE + head * SL;

    float f[4];
    float m = -1e30f, ssum = 0.0f;
#pragma unroll
    for (int j = 0; j < 4; ++j) {
        int s = threadIdx.x + j * 256;
        float v = Sh[s] - cl[s];
        f[j] = v;
        float nm = fmaxf(m, v);
        ssum = ssum * __expf(m - nm) + __expf(v - nm);
        m = nm;
    }
    __shared__ float sm[256];
    __shared__ float ss[256];
    sm[threadIdx.x] = m;
    ss[threadIdx.x] = ssum;
    __syncthreads();
    for (int off = 128; off > 0; off >>= 1) {
        if ((int)threadIdx.x < off) {
            float m1 = sm[threadIdx.x], s1 = ss[threadIdx.x];
            float m2 = sm[threadIdx.x + off], s2 = ss[threadIdx.x + off];
            float nm = fmaxf(m1, m2);
            sm[threadIdx.x] = nm;
            ss[threadIdx.x] = s1 * __expf(m1 - nm) + s2 * __expf(m2 - nm);
        }
        __syncthreads();
    }
    float lse = sm[0] + __logf(ss[0]);
#pragma unroll
    for (int j = 0; j < 4; ++j) {
        int s = threadIdx.x + j * 256;
        float v = f[j] - lse;
        Sh[s] = final_exp ? __expf(v) : v;
    }
}

// ---------------------------------------------------------------------------
// Stage 4: O = P @ V per head.  A = f32 probs (converted in loader),
// Bt = vt[head].  2x2 blocking, direct loads (S streams through L2).
// ---------------------------------------------------------------------------
__global__ void __launch_bounds__(256) av_gemm(
    const float* __restrict__ S, const _Float16* __restrict__ vt,
    _Float16* __restrict__ ocat) {
    int lane = threadIdx.x & 31, wv = threadIdx.x >> 5;
    int head = blockIdx.z;
    int m0 = blockIdx.y * 128 + (wv >> 1) * 32;   // l tile
    int n0 = (wv & 1) * 32;                       // d tile (N=64 total)
    const float*    Sh = S  + (size_t)head * SL * SL;
    const _Float16* vb = vt + (size_t)head * SD * SL;
    v8f acc[2][2] = {};
    for (int k = 0; k < SL; k += 32) {
        __builtin_prefetch(Sh + (size_t)m0 * SL + k + 64, 0, 1);
        v16h a0 = load_fragA_f32src(Sh, SL, m0, k, lane);
        v16h a1 = load_fragA_f32src(Sh, SL, m0 + 16, k, lane);
        v16h b0 = load_fragB_f16(vb, SL, n0, k, lane);
        v16h b1 = load_fragB_f16(vb, SL, n0 + 16, k, lane);
        acc[0][0] = WMMA_F16(a0, b0, acc[0][0]);
        acc[0][1] = WMMA_F16(a0, b1, acc[0][1]);
        acc[1][0] = WMMA_F16(a1, b0, acc[1][0]);
        acc[1][1] = WMMA_F16(a1, b1, acc[1][1]);
    }
    int b_ = head >> 4;   // batch
    int h  = head & 15;   // head index
    int g = lane >> 4, c = lane & 15;
#pragma unroll
    for (int tm = 0; tm < 2; ++tm)
#pragma unroll
        for (int tn = 0; tn < 2; ++tn) {
            int d = n0 + tn * 16 + c;
#pragma unroll
            for (int i = 0; i < 8; ++i) {
                int l = m0 + tm * 16 + g * 8 + i;
                ocat[((size_t)l * SB + b_) * SE + h * SD + d] =
                    (_Float16)acc[tm][tn][i];
            }
        }
}

// ---------------------------------------------------------------------------
// Stage 5: out projection, LDS-staged.  out = ocat @ woh^T + bias (f32).
// ---------------------------------------------------------------------------
__global__ void __launch_bounds__(256) out_gemm(
    const _Float16* __restrict__ ocat, const _Float16* __restrict__ woh,
    const float* __restrict__ bias, float* __restrict__ out) {
    __shared__ _Float16 ldsA[2][128 * LDS_PAD];
    __shared__ _Float16 ldsB[2][64 * LDS_PAD];
    int mblk = blockIdx.y * 128;
    int nblk = blockIdx.x * 64;
    v8f acc[2][2] = {};
    gemm_core_staged(ocat, woh, SE, SE, SE, mblk, nblk, ldsA, ldsB, acc);

    int lane = threadIdx.x & 31, wv = threadIdx.x >> 5;
    int g = lane >> 4, c = lane & 15;
    int mw = mblk + (wv >> 1) * 32;
    int nw = nblk + (wv & 1) * 32;
#pragma unroll
    for (int tm = 0; tm < 2; ++tm)
#pragma unroll
        for (int tn = 0; tn < 2; ++tn) {
            int n = nw + tn * 16 + c;
            float bj = bias[n];
#pragma unroll
            for (int i = 0; i < 8; ++i) {
                int m = mw + tm * 16 + g * 8 + i;
                out[(size_t)m * SE + n] = acc[tm][tn][i] + bj;
            }
        }
}

// ---------------------------------------------------------------------------
// Host-side launch sequence (all on `stream`, graph-capture safe).
// ---------------------------------------------------------------------------
extern "C" void kernel_launch(void* const* d_in, const int* in_sizes, int n_in,
                              void* d_out, int out_size, void* d_ws, size_t ws_size,
                              hipStream_t stream) {
    const float* x     = (const float*)d_in[0];   // [L,B,E]
    const float* in_w  = (const float*)d_in[1];   // [3E,E]
    const float* in_b  = (const float*)d_in[2];   // [3E]
    const float* out_w = (const float*)d_in[3];   // [E,E]
    const float* out_b = (const float*)d_in[4];   // [E]
    float* out = (float*)d_out;                   // [L,B,E] f32

    char* p = (char*)d_ws;
    auto alloc = [&](size_t bytes) -> void* {
        void* r = (void*)p;
        p += (bytes + 255) & ~(size_t)255;
        return r;
    };
    _Float16* xh   = (_Float16*)alloc((size_t)SM * SE * 2);            // 16 MB
    _Float16* wih  = (_Float16*)alloc((size_t)3 * SE * SE * 2);        //  6 MB
    _Float16* woh  = (_Float16*)alloc((size_t)SE * SE * 2);            //  2 MB
    _Float16* qh   = (_Float16*)alloc((size_t)SB * SH * SL * SD * 2);  // 16 MB
    _Float16* kh   = (_Float16*)alloc((size_t)SB * SH * SL * SD * 2);  // 16 MB
    _Float16* vt   = (_Float16*)alloc((size_t)SB * SH * SD * SL * 2);  // 16 MB
    float*    S    = (float*)   alloc((size_t)SB * SH * SL * SL * 4);  // 512 MB
    float*    clse = (float*)   alloc((size_t)SB * SH * SL * 4);       // 0.5 MB
    _Float16* ocat = (_Float16*)alloc((size_t)SM * SE * 2);            // 16 MB

    // Stage 0: precision downconvert
    cvt_f32_to_f16<<<1024, 256, 0, stream>>>(x,     xh,  SM * SE);
    cvt_f32_to_f16<<<1024, 256, 0, stream>>>(in_w,  wih, 3 * SE * SE);
    cvt_f32_to_f16<<<1024, 256, 0, stream>>>(out_w, woh, SE * SE);

    // Stage 1: QKV projection (M=8192, N=3072, K=1024), block tile 128x64
    qkv_gemm<<<dim3(3 * SE / 64, SM / 128), 256, 0, stream>>>(
        xh, wih, in_b, qh, kh, vt);

    // Stage 2: scores per head (M=N=1024, K=64)
    scores_gemm<<<dim3(SL / 64, SL / 128, SB * SH), 256, 0, stream>>>(qh, kh, S);

    // Stage 3: log-domain Sinkhorn, 3 iterations (final row pass emits exp)
    for (int it = 0; it < 3; ++it) {
        sinkhorn_col<<<dim3(SL / 256, SB * SH), 256, 0, stream>>>(S, clse);
        sinkhorn_row<<<dim3(SL, SB * SH), 256, 0, stream>>>(S, clse, it == 2);
    }

    // Stage 4: O = P @ V per head (M=1024, N=64, K=1024)
    av_gemm<<<dim3(1, SL / 128, SB * SH), 256, 0, stream>>>(S, vt, ocat);

    // Stage 5: output projection (M=8192, N=1024, K=1024) + bias -> f32
    out_gemm<<<dim3(SE / 64, SM / 128), 256, 0, stream>>>(ocat, woh, out_b, out);
}
